// SelfAttention_7507602833797
// MI455X (gfx1250) — matrix operand order repaired and checked by reading, not compiled, not run on hardware
//
#include <hip/hip_runtime.h>
#include <hip/hip_bf16.h>

// ---------------------------------------------------------------------------
// Problem constants (reference: B=2, S=2048, H=2048, NH=16, HD=128)
// ---------------------------------------------------------------------------
#define Bc  2
#define Sc  2048
#define Hc  2048
#define NHc 16
#define HDc 128

typedef __attribute__((ext_vector_type(16))) __bf16 v16bf;
typedef __attribute__((ext_vector_type(8)))  float  v8f;

union AFrag {
    v16bf v;
    unsigned short u[16];
    unsigned int   d[8];
};

__device__ __forceinline__ unsigned short f2bf(float f) {
    unsigned int u = __float_as_uint(f);
    unsigned int r = u + 0x7FFFu + ((u >> 16) & 1u);   // round-to-nearest-even
    return (unsigned short)(r >> 16);
}

// ---------------------------------------------------------------------------
// CDNA5 async global->LDS copy (GLOBAL_LOAD_ASYNC_TO_LDS_B128, ASYNCcnt).
// ---------------------------------------------------------------------------
#if __has_builtin(__builtin_amdgcn_global_load_async_to_lds_b128)
#define HAVE_ASYNC_LDS 1
#else
#define HAVE_ASYNC_LDS 0
#endif

typedef int v4i __attribute__((vector_size(16)));
typedef __attribute__((address_space(1))) v4i* gptr4;
typedef __attribute__((address_space(3))) v4i* lptr4;

__device__ __forceinline__ void copy16_g2l(const unsigned short* g, unsigned short* l)
{
#if HAVE_ASYNC_LDS
    __builtin_amdgcn_global_load_async_to_lds_b128(
        (gptr4)(unsigned long long)(uintptr_t)g,
        (lptr4)(unsigned int)(uintptr_t)l,
        0, 0);
#else
    *(uint4*)l = *(const uint4*)g;
#endif
}

__device__ __forceinline__ void async_wait_all()
{
#if HAVE_ASYNC_LDS
#if __has_builtin(__builtin_amdgcn_s_wait_asynccnt)
    __builtin_amdgcn_s_wait_asynccnt(0);
#else
    asm volatile("s_wait_asynccnt 0" ::: "memory");
#endif
#endif
}

// ---------------------------------------------------------------------------
// fp32 -> bf16 elementwise convert
// ---------------------------------------------------------------------------
__global__ __launch_bounds__(256) void cvt_f32_bf16_kernel(
    const float* __restrict__ in, unsigned short* __restrict__ out, int n)
{
    int i = blockIdx.x * 256 + threadIdx.x;
    if (i < n) out[i] = f2bf(in[i]);
}

// ---------------------------------------------------------------------------
// C[M,N] = A[M,K](bf16) @ W[N,K]^T(bf16) + bias   (torch Linear convention)
// Block: 256 threads (8 waves, 2x4). Block tile 128x256, K-step 32.
// Wave tile 64x64: 4 A-frags + 4 B-frags -> 16 WMMAs (1.0 LDS load / WMMA).
// Double-buffered LDS; tile k+1 streams in via async DMA during compute.
// ---------------------------------------------------------------------------
__global__ __launch_bounds__(256) void gemm_bf16_wmma_kernel(
    const unsigned short* __restrict__ A,
    const unsigned short* __restrict__ W,
    const float* __restrict__ bias,
    unsigned short* __restrict__ outB,   // bf16 out (or null)
    float* __restrict__ outF,            // f32 out (or null)
    int M, int N, int K)
{
    __shared__ __align__(16) unsigned short As[2][128][32];
    __shared__ __align__(16) unsigned short Ws[2][256][32];

    const int tid   = threadIdx.x;
    const int lane  = tid & 31;
    const int wave  = tid >> 5;
    const int l16   = lane & 15;
    const int half  = lane >> 4;      // lane half selects K sub-range / M+8
    const int waveM = wave >> 2;      // 0..1  -> 64-row band
    const int waveN = wave & 3;       // 0..3  -> 64-col band
    const int m0    = blockIdx.x * 128;
    const int n0    = blockIdx.y * 256;

    // per-thread DMA chunks: A tile 512 chunks (2/thread), W tile 1024 (4/thread)
    const int ra = tid >> 2;               // 0..63
    const int ca = (tid & 3) * 8;          // elem offset in 32-elem row

    v8f zero = {};
    v8f acc[4][4];
    #pragma unroll
    for (int mi = 0; mi < 4; ++mi)
        #pragma unroll
        for (int ni = 0; ni < 4; ++ni) acc[mi][ni] = zero;

    // prologue: stream tile 0 into buffer 0
    {
        copy16_g2l(A + (size_t)(m0 + ra) * K + ca,      &As[0][ra][ca]);
        copy16_g2l(A + (size_t)(m0 + ra + 64) * K + ca, &As[0][ra + 64][ca]);
        #pragma unroll
        for (int i = 0; i < 4; ++i) {
            const int idx = tid + i * 256;
            const int r = idx >> 2, c8 = (idx & 3) * 8;
            copy16_g2l(W + (size_t)(n0 + r) * K + c8, &Ws[0][r][c8]);
        }
    }

    int cur = 0;
    for (int k0 = 0; k0 < K; k0 += 32) {
        async_wait_all();          // my DMA into buf[cur] landed
        __syncthreads();           // everyone's DMA into buf[cur] landed

        // stream tile k+1 into the other buffer while we compute
        if (k0 + 32 < K) {
            const int nxt = cur ^ 1;
            const int kn  = k0 + 32;
            copy16_g2l(A + (size_t)(m0 + ra) * K + kn + ca,      &As[nxt][ra][ca]);
            copy16_g2l(A + (size_t)(m0 + ra + 64) * K + kn + ca, &As[nxt][ra + 64][ca]);
            #pragma unroll
            for (int i = 0; i < 4; ++i) {
                const int idx = tid + i * 256;
                const int r = idx >> 2, c8 = (idx & 3) * 8;
                copy16_g2l(W + (size_t)(n0 + r) * K + kn + c8, &Ws[nxt][r][c8]);
            }
        }

        // ---- build fragments (ISA 7.12.2 layouts)
        AFrag af[4], wf[4];
        #pragma unroll
        for (int mi = 0; mi < 4; ++mi) {
            const unsigned int* Ar =
                (const unsigned int*)&As[cur][waveM * 64 + mi * 16 + l16][0];
            #pragma unroll
            for (int v = 0; v < 4; ++v) {
                af[mi].d[v]     = Ar[half * 4 + v];          // K = half*8 + 2v
                af[mi].d[4 + v] = Ar[half * 4 + 8 + v];      // K = half*8 + 16 + 2v
            }
        }
        #pragma unroll
        for (int ni = 0; ni < 4; ++ni) {
            const unsigned int* Wr =
                (const unsigned int*)&Ws[cur][waveN * 64 + ni * 16 + l16][0];
            #pragma unroll
            for (int v = 0; v < 4; ++v) {
                wf[ni].d[v]     = Wr[half * 4 + v];          // K = half*8 + 2v
                wf[ni].d[4 + v] = Wr[half * 4 + 8 + v];      // K = half*8 + 16 + 2v
            }
        }

        // ---- 16 WMMAs per K-step (64x64 wave tile)
        #pragma unroll
        for (int mi = 0; mi < 4; ++mi)
            #pragma unroll
            for (int ni = 0; ni < 4; ++ni)
                acc[mi][ni] = __builtin_amdgcn_wmma_f32_16x16x32_bf16(
                    false, af[mi].v, false, wf[ni].v,
                    (short)0, acc[mi][ni], false, false);

        __syncthreads();           // all reads of buf[cur] done before overwrite
        cur ^= 1;
    }

    // ---- epilogue: bias add, store (C/D layout: N=l16, M=v + 8*half)
    #pragma unroll
    for (int mi = 0; mi < 4; ++mi) {
        #pragma unroll
        for (int ni = 0; ni < 4; ++ni) {
            const int n  = n0 + waveN * 64 + ni * 16 + l16;
            const float bv = bias[n];
            #pragma unroll
            for (int v = 0; v < 8; ++v) {
                const int m = m0 + waveM * 64 + mi * 16 + v + half * 8;
                const float c = acc[mi][ni][v] + bv;
                if (outF) outF[(size_t)m * N + n] = c;
                else      outB[(size_t)m * N + n] = f2bf(c);
            }
        }
    }
}

// ---------------------------------------------------------------------------
// Causal flash attention over bf16 Q/K/V stored as [B, S, H] (heads packed).
// Block: 128 threads (4 waves); each wave owns 16 query rows; block = 64 rows.
// Key tiles of 32. K tile streams in via async DMA; V tile is register-staged
// and transposed into LDS (B-operand layout needs [d][key]).
// ---------------------------------------------------------------------------
__global__ __launch_bounds__(128) void attn_flash_wmma_kernel(
    const unsigned short* __restrict__ Q,
    const unsigned short* __restrict__ Kg,
    const unsigned short* __restrict__ Vg,
    unsigned short* __restrict__ O,
    int S, int H, int NH)
{
    __shared__ __align__(16) unsigned short Ks[32][128];   // [key][d]
    __shared__ __align__(16) unsigned short Vt[128][32];   // [d][key]
    __shared__ __align__(16) unsigned short Ps[4][16][32]; // per-wave P tile

    const int tid  = threadIdx.x;
    const int lane = tid & 31;
    const int wave = tid >> 5;
    const int l16  = lane & 15;
    const int half = lane >> 4;

    const int qblocks = S >> 6;                       // 64 q rows per block
    const int qblk    = blockIdx.x % qblocks;
    const int head    = (blockIdx.x / qblocks) % NH;
    const int batch   =  blockIdx.x / (qblocks * NH);

    const int q0 = qblk * 64 + wave * 16;
    const size_t base = ((size_t)batch * S) * H + (size_t)head * HDc;

    // ---- Q fragments: A-layout, 16 rows x 128 d, 4 K-steps of 32
    AFrag qf[4];
    {
        const unsigned int* Qrow =
            (const unsigned int*)(Q + base + (size_t)(q0 + l16) * H); // 64 dwords
        #pragma unroll
        for (int f = 0; f < 4; ++f)
            #pragma unroll
            for (int v = 0; v < 4; ++v) {
                qf[f].d[v]     = Qrow[f * 16 + half * 4 + v];
                qf[f].d[4 + v] = Qrow[f * 16 + half * 4 + 8 + v];
            }
    }

    v8f zero = {};
    v8f oacc[8];
    #pragma unroll
    for (int dt = 0; dt < 8; ++dt) oacc[dt] = zero;
    float mrow[8], lrow[8];
    #pragma unroll
    for (int v = 0; v < 8; ++v) { mrow[v] = -1e30f; lrow[v] = 0.f; }

    const float sscale = 0.08838834764831845f;   // 1/sqrt(128)
    const int ktiles = qblk * 2 + 2;             // keys <= last q row of block
    const int qmax   = q0 + 15;

    for (int kt = 0; kt < ktiles; ++kt) {
        const int kk0 = kt * 32;
        __syncthreads();           // previous compute done; tiles reusable

        // ---- K tile: 32x128 bf16 = 512 x 16B chunks, async DMA, 4/thread
        #pragma unroll
        for (int i = 0; i < 4; ++i) {
            const int idx = tid + i * 128;       // 0..511
            const int row = idx >> 4;            // key 0..31
            const int c8  = (idx & 15) * 8;      // elem offset in 128-elem row
            copy16_g2l(Kg + base + (size_t)(kk0 + row) * H + c8, &Ks[row][c8]);
        }

        // ---- V tile: register-stage 16 dwords, then transposed LDS stores
        unsigned int vtmp[16];
        #pragma unroll
        for (int j = 0; j < 16; ++j) {
            const int idx = tid + j * 128;       // 0..2047 dwords
            const int row = idx >> 6;            // key 0..31
            const int c   = idx & 63;            // dword in 128-elem row
            vtmp[j] = *(const unsigned int*)(Vg + base + (size_t)(kk0 + row) * H
                                             + (size_t)c * 2);
        }
        #pragma unroll
        for (int j = 0; j < 16; ++j) {
            const int idx = tid + j * 128;
            const int row = idx >> 6;
            const int c   = idx & 63;
            Vt[c * 2    ][row] = (unsigned short)(vtmp[j] & 0xffffu);
            Vt[c * 2 + 1][row] = (unsigned short)(vtmp[j] >> 16);
        }

        async_wait_all();
        __syncthreads();
        if (kk0 > qmax) continue;                // barrier counts stay uniform

        // ---- scores: S = Q K^T (two 16-key subtiles, K-dim = d = 128)
        v8f s[2];
        s[0] = zero; s[1] = zero;
        #pragma unroll
        for (int sub = 0; sub < 2; ++sub) {
            const unsigned int* Krow =
                (const unsigned int*)&Ks[sub * 16 + l16][0];  // 64 dwords
            #pragma unroll
            for (int f = 0; f < 4; ++f) {
                AFrag kf;
                #pragma unroll
                for (int v = 0; v < 4; ++v) {
                    kf.d[v]     = Krow[f * 16 + half * 4 + v];     // d = f*32+half*8+2v
                    kf.d[4 + v] = Krow[f * 16 + half * 4 + 8 + v]; // d = f*32+16+half*8+2v
                }
                s[sub] = __builtin_amdgcn_wmma_f32_16x16x32_bf16(
                    false, qf[f].v, false, kf.v, (short)0, s[sub], false, false);
            }
        }

        // ---- scale + causal mask + row-max (reduce over 16-lane half)
        float tmax[8];
        #pragma unroll
        for (int v = 0; v < 8; ++v) {
            const int qr = q0 + v + half * 8;
            float s0 = s[0][v] * sscale;
            float s1 = s[1][v] * sscale;
            if (kk0 + l16      > qr) s0 = -1e30f;
            if (kk0 + 16 + l16 > qr) s1 = -1e30f;
            s[0][v] = s0; s[1][v] = s1;
            tmax[v] = fmaxf(s0, s1);
        }
        #pragma unroll
        for (int off = 1; off < 16; off <<= 1)
            #pragma unroll
            for (int v = 0; v < 8; ++v)
                tmax[v] = fmaxf(tmax[v], __shfl_xor(tmax[v], off, 32));

        // ---- online softmax update
        float alpha[8], lsum[8];
        #pragma unroll
        for (int v = 0; v < 8; ++v) {
            const float mn = fmaxf(mrow[v], tmax[v]);
            alpha[v] = __expf(mrow[v] - mn);
            mrow[v]  = mn;
        }
        #pragma unroll
        for (int v = 0; v < 8; ++v) {
            s[0][v] = __expf(s[0][v] - mrow[v]);
            s[1][v] = __expf(s[1][v] - mrow[v]);
            lsum[v] = s[0][v] + s[1][v];
        }
        #pragma unroll
        for (int off = 1; off < 16; off <<= 1)
            #pragma unroll
            for (int v = 0; v < 8; ++v)
                lsum[v] += __shfl_xor(lsum[v], off, 32);
        #pragma unroll
        for (int v = 0; v < 8; ++v)
            lrow[v] = lrow[v] * alpha[v] + lsum[v];
        #pragma unroll
        for (int dt = 0; dt < 8; ++dt)
            #pragma unroll
            for (int v = 0; v < 8; ++v)
                oacc[dt][v] = oacc[dt][v] * alpha[v];

        // ---- P: C-layout regs -> per-wave LDS -> A-layout fragment
        #pragma unroll
        for (int sub = 0; sub < 2; ++sub)
            #pragma unroll
            for (int v = 0; v < 8; ++v)
                Ps[wave][v + half * 8][sub * 16 + l16] = f2bf(s[sub][v]);
        asm volatile("s_wait_dscnt 0" ::: "memory");   // within-wave LDS RAW

        AFrag pf;
        {
            const unsigned int* Prow =
                (const unsigned int*)&Ps[wave][l16][0];       // 16 dwords
            #pragma unroll
            for (int v = 0; v < 4; ++v) {
                pf.d[v]     = Prow[half * 4 + v];
                pf.d[4 + v] = Prow[half * 4 + 8 + v];
            }
        }

        // ---- O += P @ V   (K-dim = 32 keys, 8 d-subtiles)
        #pragma unroll
        for (int dt = 0; dt < 8; ++dt) {
            const unsigned int* Vrow =
                (const unsigned int*)&Vt[dt * 16 + l16][0];   // 16 dwords
            AFrag vf;
            #pragma unroll
            for (int v = 0; v < 4; ++v) {
                vf.d[v]     = Vrow[half * 4 + v];             // keys half*8+2v
                vf.d[4 + v] = Vrow[half * 4 + 8 + v];         // keys 16+half*8+2v
            }
            oacc[dt] = __builtin_amdgcn_wmma_f32_16x16x32_bf16(
                false, pf.v, false, vf.v, (short)0, oacc[dt], false, false);
        }
    }

    // ---- normalize and store O (bf16, [B,S,H] with head offset)
    #pragma unroll
    for (int v = 0; v < 8; ++v) lrow[v] = 1.0f / lrow[v];
    #pragma unroll
    for (int dt = 0; dt < 8; ++dt)
        #pragma unroll
        for (int v = 0; v < 8; ++v) {
            const int qq = q0 + v + half * 8;
            const int d  = dt * 16 + l16;
            O[base + (size_t)qq * H + d] = f2bf(oacc[dt][v] * lrow[v]);
        }
}

// ---------------------------------------------------------------------------
// Host-side launch
// ---------------------------------------------------------------------------
extern "C" void kernel_launch(void* const* d_in, const int* in_sizes, int n_in,
                              void* d_out, int out_size, void* d_ws, size_t ws_size,
                              hipStream_t stream) {
    (void)in_sizes; (void)n_in; (void)out_size; (void)ws_size;

    const float* x   = (const float*)d_in[0];
    const float* WQw = (const float*)d_in[1];
    const float* WQb = (const float*)d_in[2];
    const float* WKw = (const float*)d_in[3];
    const float* WKb = (const float*)d_in[4];
    const float* WVw = (const float*)d_in[5];
    const float* WVb = (const float*)d_in[6];
    const float* WOw = (const float*)d_in[7];
    const float* WOb = (const float*)d_in[8];

    const int M  = Bc * Sc;          // 4096 rows
    const int H  = Hc;
    const size_t nX = (size_t)M * H;       // 8.39M
    const size_t nW = (size_t)H * H;       // 4.19M

    // workspace carve-up (ushort units); total ~112 MB
    unsigned short* ws = (unsigned short*)d_ws;
    size_t off = 0;
    unsigned short* xb  = ws + off; off += nX;
    unsigned short* wqb = ws + off; off += nW;
    unsigned short* wkb = ws + off; off += nW;
    unsigned short* wvb = ws + off; off += nW;
    unsigned short* wob = ws + off; off += nW;
    unsigned short* Qb  = ws + off; off += nX;
    unsigned short* Kb  = ws + off; off += nX;
    unsigned short* Vb  = ws + off; off += nX;
    unsigned short* Ob  = ws + off; off += nX;

    // 1) fp32 -> bf16 converts
    {
        int gx = (int)((nX + 255) / 256);
        cvt_f32_bf16_kernel<<<gx, 256, 0, stream>>>(x, xb, (int)nX);
        int gw = (int)((nW + 255) / 256);
        cvt_f32_bf16_kernel<<<gw, 256, 0, stream>>>(WQw, wqb, (int)nW);
        cvt_f32_bf16_kernel<<<gw, 256, 0, stream>>>(WKw, wkb, (int)nW);
        cvt_f32_bf16_kernel<<<gw, 256, 0, stream>>>(WVw, wvb, (int)nW);
        cvt_f32_bf16_kernel<<<gw, 256, 0, stream>>>(WOw, wob, (int)nW);
    }

    // 2) QKV projections: [M,H] = xb @ W^T + b  (bf16 out)
    dim3 gGrid(M / 128, H / 256);
    gemm_bf16_wmma_kernel<<<gGrid, 256, 0, stream>>>(xb, wqb, WQb, Qb, nullptr, M, H, H);
    gemm_bf16_wmma_kernel<<<gGrid, 256, 0, stream>>>(xb, wkb, WKb, Kb, nullptr, M, H, H);
    gemm_bf16_wmma_kernel<<<gGrid, 256, 0, stream>>>(xb, wvb, WVb, Vb, nullptr, M, H, H);

    // 3) causal flash attention: 64 q-rows per block
    {
        int nblk = Bc * NHc * (Sc / 64);   // 1024
        attn_flash_wmma_kernel<<<nblk, 128, 0, stream>>>(Qb, Kb, Vb, Ob, Sc, H, NHc);
    }

    // 4) output projection: d_out(f32) = Ob @ WO^T + b
    gemm_bf16_wmma_kernel<<<gGrid, 256, 0, stream>>>(Ob, wob, WOb, nullptr,
                                                     (float*)d_out, M, H, H);
}
